// IlluminationBranch_28106265985463
// MI455X (gfx1250) — compile-verified, matrix operand order checked by reading, not run on hardware
//
#include <hip/hip_runtime.h>

// ---------------- CDNA5 (gfx1250, wave32) types ----------------
typedef __attribute__((ext_vector_type(16))) _Float16 half16;
typedef __attribute__((ext_vector_type(8)))  _Float16 half8;
typedef __attribute__((ext_vector_type(8)))  float    float8;

#define DI_CH 128
#define K_DIR 4
#define XDBL_R 48   // padded rows of xdbl (36 -> 48)

__device__ __forceinline__ float silu_f(float v) { return v / (1.f + __expf(-v)); }

// =====================================================================
// All-f16-operand GEMM on the WMMA pipe.
//   A: [M, K] f16 row-major (lda = padded K), M % 16 == 0, K % 32 == 0
//   B: [N, K] f16 row-major (ldb = padded K), N % 64 == 0  (K-major = "TN")
//   Cm: [M, N] f32 (ldc);  optional CT: [N, ldct] f16 transposed copy
// One wave32 per 16(M) x 64(N) tile. Per 32-K step: 2 b128 loads for A,
// 8 b128 loads for B, then 4x v_wmma_f32_16x16x32_f16. No converts in loop.
// Fragment layout per CDNA5 ISA 7.12.2: lane p holds row m0+p (A)
// / col n (B); VGPR runs are two contiguous 8-half K-spans.
// =====================================================================
__global__ void gemm_wmma_f16(const _Float16* __restrict__ A, const _Float16* __restrict__ B,
                              float* __restrict__ Cm, _Float16* __restrict__ CT,
                              int K, int lda, int ldb, long ldc, int ldct, int writeCT,
                              long bA, long bB, long bC, long bCT)
{
  const int lane = threadIdx.x;        // one wave
  const int lh   = lane >> 4;          // K-half selector
  const int p    = lane & 15;          // A row / B,C column
  const long n0  = (long)blockIdx.x * 64;
  const long m0  = (long)blockIdx.y * 16;
  A  += (long)blockIdx.z * bA + (m0 + p) * (long)lda;
  B  += (long)blockIdx.z * bB + (n0 + p) * (long)ldb;
  Cm += (long)blockIdx.z * bC;
  CT += (long)blockIdx.z * bCT;

  float8 acc0 = {0,0,0,0,0,0,0,0}, acc1 = acc0, acc2 = acc0, acc3 = acc0;

  for (int k0 = 0; k0 < K; k0 += 32) {
    const int klo = k0 + (lh << 3);        // k0 + 8*lh
    const int khi = klo + 16;
    half8 al = *(const half8*)(A + klo);
    half8 ah = *(const half8*)(A + khi);
    half8 b0l = *(const half8*)(B + klo),            b0h = *(const half8*)(B + khi);
    half8 b1l = *(const half8*)(B + 16L*ldb + klo),  b1h = *(const half8*)(B + 16L*ldb + khi);
    half8 b2l = *(const half8*)(B + 32L*ldb + klo),  b2h = *(const half8*)(B + 32L*ldb + khi);
    half8 b3l = *(const half8*)(B + 48L*ldb + klo),  b3h = *(const half8*)(B + 48L*ldb + khi);
    if (k0 + 32 < K) {
      __builtin_prefetch(A + k0 + 32, 0, 1);                 // global_prefetch_b8
      __builtin_prefetch(B + k0 + 32, 0, 1);
    }
    half16 aF  = __builtin_shufflevector(al,  ah,  0,1,2,3,4,5,6,7,8,9,10,11,12,13,14,15);
    half16 b0F = __builtin_shufflevector(b0l, b0h, 0,1,2,3,4,5,6,7,8,9,10,11,12,13,14,15);
    half16 b1F = __builtin_shufflevector(b1l, b1h, 0,1,2,3,4,5,6,7,8,9,10,11,12,13,14,15);
    half16 b2F = __builtin_shufflevector(b2l, b2h, 0,1,2,3,4,5,6,7,8,9,10,11,12,13,14,15);
    half16 b3F = __builtin_shufflevector(b3l, b3h, 0,1,2,3,4,5,6,7,8,9,10,11,12,13,14,15);
    acc0 = __builtin_amdgcn_wmma_f32_16x16x32_f16(false, aF, false, b0F, (short)0, acc0, false, false);
    acc1 = __builtin_amdgcn_wmma_f32_16x16x32_f16(false, aF, false, b1F, (short)0, acc1, false, false);
    acc2 = __builtin_amdgcn_wmma_f32_16x16x32_f16(false, aF, false, b2F, (short)0, acc2, false, false);
    acc3 = __builtin_amdgcn_wmma_f32_16x16x32_f16(false, aF, false, b3F, (short)0, acc3, false, false);
  }

#pragma unroll
  for (int v = 0; v < 8; ++v) {
    // C/D 16x16 f32 layout: lanes 0-15 -> M=v, lanes 16-31 -> M=v+8
    float* cp = Cm + (m0 + (lh << 3) + v) * ldc + n0 + p;
    cp[0]  = acc0[v];
    cp[16] = acc1[v];
    cp[32] = acc2[v];
    cp[48] = acc3[v];
  }
  if (writeCT) {
    // CT[n, m]: lane's 8 acc values are M-contiguous -> one half8 store each
    half8 h0, h1, h2, h3;
#pragma unroll
    for (int v = 0; v < 8; ++v) {
      h0[v] = (_Float16)acc0[v];
      h1[v] = (_Float16)acc1[v];
      h2[v] = (_Float16)acc2[v];
      h3[v] = (_Float16)acc3[v];
    }
    const long mo = m0 + (lh << 3);
    *(half8*)(CT + (n0 + p)      * (long)ldct + mo) = h0;
    *(half8*)(CT + (n0 + p + 16) * (long)ldct + mo) = h1;
    *(half8*)(CT + (n0 + p + 32) * (long)ldct + mo) = h2;
    *(half8*)(CT + (n0 + p + 48) * (long)ldct + mo) = h3;
  }
}

// Pack f32 weights to padded f16 [batch, pM, pK]; t=1 transposes (src is [K, M]).
__global__ void pack_f16(const float* __restrict__ src, _Float16* __restrict__ dst,
                         int M, int Kd, int pM, int pK, long srcC, int t,
                         int batch, long sBatch)
{
  int idx = blockIdx.x * blockDim.x + threadIdx.x;
  int tot = batch * pM * pK;
  if (idx >= tot) return;
  int b = idx / (pM * pK);
  int m = (idx / pK) % pM;
  int k = idx % pK;
  float v = 0.f;
  if (m < M && k < Kd)
    v = t ? src[(long)b * sBatch + (long)k * srcC + m]
          : src[(long)b * sBatch + (long)m * srcC + k];
  dst[idx] = (_Float16)v;
}

// =====================================================================
// Patch embed: 4x4/4 conv of 1x1x256x256 -> [64,64,64]
// =====================================================================
__global__ void patch_embed(const float* __restrict__ img, const float* __restrict__ w,
                            const float* __restrict__ b, float* __restrict__ out)
{
  int idx = blockIdx.x * blockDim.x + threadIdx.x;
  if (idx >= 64 * 64 * 64) return;
  int c = idx >> 12;
  int r = idx & 4095;
  int ho = r >> 6, wo = r & 63;
  const float* wp = w + c * 16;
  float acc = b[c];
#pragma unroll
  for (int i = 0; i < 4; ++i)
#pragma unroll
    for (int j = 0; j < 4; ++j)
      acc += wp[i * 4 + j] * img[(ho * 4 + i) * 256 + (wo * 4 + j)];
  out[idx] = acc;
}

// LayerNorm over channels, f32 out, [Cc, HW] layout (patch embed / downsample).
__global__ void ln_channels(const float* __restrict__ in, float* __restrict__ out,
                            const float* __restrict__ g, const float* __restrict__ b,
                            int Cc, int HW)
{
  int l = blockIdx.x * blockDim.x + threadIdx.x;
  if (l >= HW) return;
  float mu = 0.f;
  for (int c = 0; c < Cc; ++c) mu += in[(long)c * HW + l];
  mu /= (float)Cc;
  float var = 0.f;
  for (int c = 0; c < Cc; ++c) { float d = in[(long)c * HW + l] - mu; var += d * d; }
  var /= (float)Cc;
  float inv = rsqrtf(var + 1e-6f);
  for (int c = 0; c < Cc; ++c)
    out[(long)c * HW + l] = (in[(long)c * HW + l] - mu) * inv * g[c] + b[c];
}

// VSS-block LayerNorm: writes f16 K-major operand hT [HW, 64] for the GEMM.
__global__ void ln_channels_f16T(const float* __restrict__ in, _Float16* __restrict__ outT,
                                 const float* __restrict__ g, const float* __restrict__ b,
                                 int HW)
{
  int l = blockIdx.x * blockDim.x + threadIdx.x;
  if (l >= HW) return;
  float mu = 0.f;
  for (int c = 0; c < 64; ++c) mu += in[(long)c * HW + l];
  mu /= 64.f;
  float var = 0.f;
  for (int c = 0; c < 64; ++c) { float d = in[(long)c * HW + l] - mu; var += d * d; }
  var /= 64.f;
  float inv = rsqrtf(var + 1e-6f);
  _Float16* row = outT + (long)l * 64;
  for (int c = 0; c < 64; ++c)
    row[c] = (_Float16)((in[(long)c * HW + l] - mu) * inv * g[c] + b[c]);
}

// Depthwise 3x3 pad 1 + bias + SiLU on the first DI channels of xz.
__global__ void dwconv3x3_silu(const float* __restrict__ in, const float* __restrict__ w,
                               const float* __restrict__ b, float* __restrict__ out,
                               int H, int W)
{
  int L = H * W;
  int idx = blockIdx.x * blockDim.x + threadIdx.x;
  if (idx >= DI_CH * L) return;
  int d = idx / L, l = idx % L;
  int hh = l / W, ww = l % W;
  const float* wp = w + d * 9;
  const float* ip = in + (long)d * L;
  float acc = b[d];
#pragma unroll
  for (int i = -1; i <= 1; ++i) {
    int h2 = hh + i;
    if (h2 < 0 || h2 >= H) continue;
#pragma unroll
    for (int j = -1; j <= 1; ++j) {
      int w2 = ww + j;
      if (w2 < 0 || w2 >= W) continue;
      acc += wp[(i + 1) * 3 + (j + 1)] * ip[h2 * W + w2];
    }
  }
  out[idx] = silu_f(acc);
}

// Scatter xc into 4 directions: f32 xs[4][DI][L] (scan) + f16 xsT[4][L][DI] (GEMM B)
__global__ void build_xs(const float* __restrict__ xc, float* __restrict__ xs,
                         _Float16* __restrict__ xsT, int H, int W)
{
  int L = H * W;
  int idx = blockIdx.x * blockDim.x + threadIdx.x;
  if (idx >= DI_CH * L) return;
  int d = idx / L, l = idx % L;
  int hh = l / W, ww = l % W;
  int l2 = ww * H + hh;                       // HxW transpose
  float v = xc[idx];
  _Float16 hv = (_Float16)v;
  long KDL = (long)DI_CH * L;
  long base = (long)d * L;
  xs[0 * KDL + base + l]            = v;
  xs[1 * KDL + base + l2]           = v;
  xs[2 * KDL + base + (L - 1 - l)]  = v;
  xs[3 * KDL + base + (L - 1 - l2)] = v;
  xsT[0 * KDL + (long)l            * DI_CH + d] = hv;
  xsT[1 * KDL + (long)l2           * DI_CH + d] = hv;
  xsT[2 * KDL + (long)(L - 1 - l)  * DI_CH + d] = hv;
  xsT[3 * KDL + (long)(L - 1 - l2) * DI_CH + d] = hv;
}

// delta = softplus(delta_pre + dt_b), layout [K,DI,L]
__global__ void softplus_bias(float* __restrict__ delta, const float* __restrict__ dtb, int L)
{
  int idx = blockIdx.x * blockDim.x + threadIdx.x;
  if (idx >= K_DIR * DI_CH * L) return;
  int k = idx / (DI_CH * L);
  int d = (idx / L) % DI_CH;
  float v = delta[idx] + dtb[k * DI_CH + d];
  delta[idx] = (v > 20.f) ? v : log1pf(__expf(v));
}

// Selective scan: one LANE per (k,d,n) state; 16-lane __shfl_xor tree reduces
// the N=16 states, lane n==0 writes y.
__global__ void selective_scan(const float* __restrict__ xs, const float* __restrict__ delta,
                               const float* __restrict__ xdbl, const float* __restrict__ A_log,
                               const float* __restrict__ Dp, float* __restrict__ ys, int L)
{
  int gid = blockIdx.x * blockDim.x + threadIdx.x;
  int seq = gid >> 4;                 // (k,d)
  int n   = gid & 15;                 // state index
  if (seq >= K_DIR * DI_CH) return;
  int k = seq / DI_CH, d = seq % DI_CH;
  const float* u  = xs    + ((long)k * DI_CH + d) * L;
  const float* dl = delta + ((long)k * DI_CH + d) * L;
  const float* Bn = xdbl + (long)k * XDBL_R * L + (long)(4 + n) * L;   // row 4+n
  const float* Cn = xdbl + (long)k * XDBL_R * L + (long)(20 + n) * L;  // row 20+n
  float Aval = -__expf(A_log[((long)k * DI_CH + d) * 16 + n]);
  float Dv = Dp[(long)k * DI_CH + d];
  float h = 0.f;
  float* yo = ys + ((long)k * DI_CH + d) * L;
  for (int l = 0; l < L; ++l) {
    float dt = dl[l], ul = u[l];
    h = __expf(dt * Aval) * h + dt * Bn[l] * ul;
    float acc = Cn[l] * h;
    acc += __shfl_xor(acc, 1, 32);
    acc += __shfl_xor(acc, 2, 32);
    acc += __shfl_xor(acc, 4, 32);
    acc += __shfl_xor(acc, 8, 32);
    if (n == 0) yo[l] = acc + Dv * ul;
  }
}

// Combine the 4 direction outputs back to [DI, H, W] (f32)
__global__ void merge_ys(const float* __restrict__ ys, float* __restrict__ y, int H, int W)
{
  int L = H * W;
  int idx = blockIdx.x * blockDim.x + threadIdx.x;
  if (idx >= DI_CH * L) return;
  int d = idx / L, l = idx % L;
  int hh = l / W, ww = l % W;
  int l2 = ww * H + hh;
  long KDL = (long)DI_CH * L;
  long base = (long)d * L;
  y[idx] = ys[0 * KDL + base + l] +
           ys[1 * KDL + base + l2] +
           ys[2 * KDL + base + (L - 1 - l)] +
           ys[3 * KDL + base + (L - 1 - l2)];
}

// Out-LN over DI per position + silu(z) gate; writes f16 K-major ytT [HW, 128].
__global__ void outnorm_gate(const float* __restrict__ y, const float* __restrict__ z,
                             _Float16* __restrict__ ytT,
                             const float* __restrict__ g, const float* __restrict__ b, int HW)
{
  int l = blockIdx.x * blockDim.x + threadIdx.x;
  if (l >= HW) return;
  float mu = 0.f;
  for (int d = 0; d < DI_CH; ++d) mu += y[(long)d * HW + l];
  mu /= (float)DI_CH;
  float var = 0.f;
  for (int d = 0; d < DI_CH; ++d) { float t = y[(long)d * HW + l] - mu; var += t * t; }
  var /= (float)DI_CH;
  float inv = rsqrtf(var + 1e-6f);
  _Float16* row = ytT + (long)l * DI_CH;
  for (int d = 0; d < DI_CH; ++d) {
    float t = (y[(long)d * HW + l] - mu) * inv * g[d] + b[d];
    float zv = z[(long)d * HW + l];
    row[d] = (_Float16)(t * silu_f(zv));
  }
}

// General 3x3 conv, pad 1, stride {1,2}, optional SiLU.
__global__ void conv3x3(const float* __restrict__ in, const float* __restrict__ w,
                        const float* __restrict__ b, float* __restrict__ out,
                        int Cin, int Cout, int H, int W, int stride, int act)
{
  int Ho = (H - 1) / stride + 1;
  int Wo = (W - 1) / stride + 1;
  int idx = blockIdx.x * blockDim.x + threadIdx.x;
  if (idx >= Cout * Ho * Wo) return;
  int co = idx / (Ho * Wo);
  int r = idx % (Ho * Wo);
  int ho = r / Wo, wo = r % Wo;
  float acc = b[co];
  for (int ci = 0; ci < Cin; ++ci) {
    const float* ip = in + (long)ci * H * W;
    const float* wp = w + ((long)co * Cin + ci) * 9;
#pragma unroll
    for (int i = 0; i < 3; ++i) {
      int hi = ho * stride + i - 1;
      if (hi < 0 || hi >= H) continue;
#pragma unroll
      for (int j = 0; j < 3; ++j) {
        int wi = wo * stride + j - 1;
        if (wi < 0 || wi >= W) continue;
        acc += wp[i * 3 + j] * ip[hi * W + wi];
      }
    }
  }
  out[idx] = act ? silu_f(acc) : acc;
}

// Half-pixel bilinear resize
__global__ void resize_bilinear(const float* __restrict__ in, float* __restrict__ out,
                                int Cc, int Hi, int Wi, int Ho, int Wo)
{
  int idx = blockIdx.x * blockDim.x + threadIdx.x;
  if (idx >= Cc * Ho * Wo) return;
  int c = idx / (Ho * Wo);
  int r = idx % (Ho * Wo);
  int ho = r / Wo, wo = r % Wo;
  float sy = (ho + 0.5f) * (float)Hi / (float)Ho - 0.5f;
  float sx = (wo + 0.5f) * (float)Wi / (float)Wo - 0.5f;
  int y0 = (int)floorf(sy), x0 = (int)floorf(sx);
  float fy = sy - (float)y0, fx = sx - (float)x0;
  int y0c = y0 < 0 ? 0 : (y0 > Hi - 1 ? Hi - 1 : y0);
  int y1 = y0 + 1;
  int y1c = y1 < 0 ? 0 : (y1 > Hi - 1 ? Hi - 1 : y1);
  int x0c = x0 < 0 ? 0 : (x0 > Wi - 1 ? Wi - 1 : x0);
  int x1 = x0 + 1;
  int x1c = x1 < 0 ? 0 : (x1 > Wi - 1 ? Wi - 1 : x1);
  const float* ip = in + (long)c * Hi * Wi;
  float v00 = ip[y0c * Wi + x0c], v01 = ip[y0c * Wi + x1c];
  float v10 = ip[y1c * Wi + x0c], v11 = ip[y1c * Wi + x1c];
  out[idx] = (1.f - fy) * ((1.f - fx) * v00 + fx * v01) +
             fy * ((1.f - fx) * v10 + fx * v11);
}

__global__ void add_inplace(float* __restrict__ dst, const float* __restrict__ src, int n)
{
  int i = blockIdx.x * blockDim.x + threadIdx.x;
  if (i < n) dst[i] += src[i];
}

// Skip-connection copy staged through LDS via the CDNA5 async path:
// global_load_async_to_lds_b32 (ASYNCcnt) + s_wait_asynccnt.
__global__ void copy_async_kernel(float* __restrict__ dst, const float* __restrict__ src, int n)
{
  __shared__ float tile[256];
  int i = blockIdx.x * blockDim.x + threadIdx.x;
  if (i < n) {
    unsigned ldsoff = (unsigned)(unsigned long long)&tile[threadIdx.x];
    unsigned long long ga = (unsigned long long)&src[i];
    asm volatile("global_load_async_to_lds_b32 %0, %1, off"
                 :: "v"(ldsoff), "v"(ga) : "memory");
  }
  asm volatile("s_wait_asynccnt 0x0" ::: "memory");
  __syncthreads();
  if (i < n) dst[i] = tile[threadIdx.x];
}

// =====================================================================
// Host orchestration
// =====================================================================
extern "C" void kernel_launch(void* const* d_in, const int* in_sizes, int n_in,
                              void* d_out, int out_size, void* d_ws, size_t ws_size,
                              hipStream_t stream)
{
  const float* E0    = (const float*)d_in[0];
  const float* pe_w  = (const float*)d_in[1];
  const float* pe_b  = (const float*)d_in[2];
  const float* pe_lg = (const float*)d_in[3];
  const float* pe_lb = (const float*)d_in[4];
  const float* ln_g  = (const float*)d_in[5];
  const float* ln_b  = (const float*)d_in[6];
  const float* in_w  = (const float*)d_in[7];
  const float* cv_w  = (const float*)d_in[8];
  const float* cv_b  = (const float*)d_in[9];
  const float* xp_w  = (const float*)d_in[10];
  const float* dt_w  = (const float*)d_in[11];
  const float* dt_b  = (const float*)d_in[12];
  const float* A_log = (const float*)d_in[13];
  const float* Dp    = (const float*)d_in[14];
  const float* on_g  = (const float*)d_in[15];
  const float* on_b  = (const float*)d_in[16];
  const float* outpw = (const float*)d_in[17];
  const float* dn_w  = (const float*)d_in[18];
  const float* dn_b  = (const float*)d_in[19];
  const float* dn_g  = (const float*)d_in[20];
  const float* dn_lb = (const float*)d_in[21];
  const float* up_w  = (const float*)d_in[22];
  const float* up_b  = (const float*)d_in[23];
  const float* oc_w  = (const float*)d_in[24];
  const float* oc_b  = (const float*)d_in[25];

  float* ws    = (float*)d_ws;
  float* x     = ws;                 // [64, 64, 64]          262144
  float* hbuf  = x     + 262144;     // conv/LN scratch       262144
  float* xz    = hbuf  + 262144;     // [256, L] f32          1048576
  float* xc    = xz    + 1048576;    // [128, L] f32          524288
  float* xs    = xc    + 524288;     // [4,128,L] f32         2097152
  float* xdbl  = xs    + 2097152;    // [4,48,L] f32          786432
  float* dbuf  = xdbl  + 786432;     // delta [4,128,L] f32   2097152
  float* ysb   = dbuf  + 2097152;    // [4,128,L] f32         2097152
  float* ytb   = ysb   + 2097152;    // [128, L] f32          524288
  float* outc  = ytb   + 524288;     // [64, L] f32           262144
  float* skip0 = outc  + 262144;     // 64x64x64              262144
  float* skip1 = skip0 + 262144;     // 64x32x32              65536
  float* skip2 = skip1 + 65536;      // 64x16x16              16384
  float* deca  = skip2 + 16384;      // decoder A             4194304
  float* decb  = deca  + 4194304;    // decoder B             262144
  // f16 operand buffers (counted in floats = halves/2), 16B aligned
  _Float16* hT      = (_Float16*)(decb + 262144);   // [L, 64]      131072 fl
  _Float16* xsT     = (_Float16*)(decb + 262144 + 131072);           // [4,L,128] 1048576 fl
  _Float16* xdblT   = (_Float16*)(decb + 262144 + 131072 + 1048576); // [4,L,48]  393216 fl
  _Float16* ytT     = (_Float16*)(decb + 262144 + 131072 + 1048576 + 393216); // [L,128] 262144 fl
  _Float16* inwPack = (_Float16*)(decb + 262144 + 131072 + 1048576 + 393216 + 262144); // [256,64] 8192 fl
  _Float16* xpPadH  = (_Float16*)(decb + 262144 + 131072 + 1048576 + 393216 + 262144 + 8192);  // [4,48,128] 12288 fl
  _Float16* dtPadH  = (_Float16*)(decb + 262144 + 131072 + 1048576 + 393216 + 262144 + 8192 + 12288); // [4,128,32] 8192 fl
  _Float16* outPack = (_Float16*)(decb + 262144 + 131072 + 1048576 + 393216 + 262144 + 8192 + 12288 + 8192); // [64,128] 4096 fl

  const int TB = 256;
#define GRID(n) dim3(((n) + TB - 1) / TB)

  // ---- patch embed + LN ----
  patch_embed<<<GRID(64 * 64 * 64), TB, 0, stream>>>(E0, pe_w, pe_b, hbuf);
  ln_channels<<<GRID(4096), TB, 0, stream>>>(hbuf, x, pe_lg, pe_lb, 64, 4096);

  float* skips[3] = { skip0, skip1, skip2 };
  const int Hs[4] = { 64, 32, 16, 8 };
  int cur = 0;
  for (int st = 0; st < 4; ++st) {
    const int H = Hs[st], W = Hs[st], L = H * W;
    for (int rep = 0; rep < 2; ++rep, ++cur) {
      // pack this block's GEMM weights to padded f16
      pack_f16<<<GRID(256 * 64), TB, 0, stream>>>(
          in_w + (long)cur * 64 * 256, inwPack, 256, 64, 256, 64, 256L, 1, 1, 0L);
      pack_f16<<<GRID(4 * 48 * 128), TB, 0, stream>>>(
          xp_w + (long)cur * 4 * 36 * 128, xpPadH, 36, 128, 48, 128, 128L, 0, 4, 36L * 128L);
      pack_f16<<<GRID(4 * 128 * 32), TB, 0, stream>>>(
          dt_w + (long)cur * 4 * 128 * 4, dtPadH, 128, 4, 128, 32, 4L, 0, 4, 512L);
      pack_f16<<<GRID(64 * 128), TB, 0, stream>>>(
          outpw + (long)cur * 128 * 64, outPack, 64, 128, 64, 128, 64L, 1, 1, 0L);
      // channel LN -> f16 K-major hT [L, 64]
      ln_channels_f16T<<<GRID(L), TB, 0, stream>>>(x, hT, ln_g + cur * 64, ln_b + cur * 64, L);
      // in-proj: xz[256, L] = inwPack[256,64] x hT[L,64]^T
      gemm_wmma_f16<<<dim3(L / 64, 16, 1), 32, 0, stream>>>(
          inwPack, hT, xz, (_Float16*)0, 64, 64, 64, (long)L, 0, 0, 0L, 0L, 0L, 0L);
      // depthwise conv + SiLU
      dwconv3x3_silu<<<GRID(DI_CH * L), TB, 0, stream>>>(
          xz, cv_w + (long)cur * DI_CH * 9, cv_b + cur * DI_CH, xc, H, W);
      // 4-direction scatter (f32 for scan + f16 K-major for GEMM)
      build_xs<<<GRID(DI_CH * L), TB, 0, stream>>>(xc, xs, xsT, H, W);
      // x-proj: xdbl[4][48,L] (+ f16 transposed xdblT[4][L,48])
      gemm_wmma_f16<<<dim3(L / 64, 3, 4), 32, 0, stream>>>(
          xpPadH, xsT, xdbl, xdblT, 128, 128, 128, (long)L, XDBL_R, 1,
          48L * 128L, (long)L * 128L, (long)XDBL_R * L, (long)L * XDBL_R);
      // delta pre-act: dbuf[4][128, L], K = 32 (zero-padded reduction)
      gemm_wmma_f16<<<dim3(L / 64, 8, 4), 32, 0, stream>>>(
          dtPadH, xdblT, dbuf, (_Float16*)0, 32, 32, XDBL_R, (long)L, 0, 0,
          128L * 32L, (long)L * XDBL_R, (long)DI_CH * L, 0L);
      // softplus(+bias)
      softplus_bias<<<GRID(4 * DI_CH * L), TB, 0, stream>>>(dbuf, dt_b + (long)cur * 4 * 128, L);
      // selective scan: lane per (k,d,n), shuffle-reduce the 16 states
      selective_scan<<<GRID(K_DIR * DI_CH * 16), TB, 0, stream>>>(
          xs, dbuf, xdbl, A_log + (long)cur * 4 * 128 * 16, Dp + (long)cur * 4 * 128, ysb, L);
      // merge directions -> ytb [128, L] f32
      merge_ys<<<GRID(DI_CH * L), TB, 0, stream>>>(ysb, ytb, H, W);
      // out-norm + silu(z) gate -> ytT [L, 128] f16
      outnorm_gate<<<GRID(L), TB, 0, stream>>>(
          ytb, xz + (long)DI_CH * L, ytT, on_g + cur * 128, on_b + cur * 128, L);
      // out-proj: outc[64, L] = outPack[64,128] x ytT[L,128]^T
      gemm_wmma_f16<<<dim3(L / 64, 4, 1), 32, 0, stream>>>(
          outPack, ytT, outc, (_Float16*)0, 128, 128, 128, (long)L, 0, 0, 0L, 0L, 0L, 0L);
      // residual
      add_inplace<<<GRID(64 * L), TB, 0, stream>>>(x, outc, 64 * L);
    }
    if (st < 3) {
      copy_async_kernel<<<GRID(64 * L), TB, 0, stream>>>(skips[st], x, 64 * L);
      const int Ho = H / 2;
      conv3x3<<<GRID(64 * Ho * Ho), TB, 0, stream>>>(
          x, dn_w + (long)st * 64 * 64 * 9, dn_b + st * 64, deca, 64, 64, H, W, 2, 0);
      ln_channels<<<GRID(Ho * Ho), TB, 0, stream>>>(
          deca, x, dn_g + st * 64, dn_lb + st * 64, 64, Ho * Ho);
    }
  }

  // ---- decoder ----
  const float* curp = x;
  int Hc = 8;
  for (int i = 0; i < 3; ++i) {
    const int Ho = Hc * 2;
    resize_bilinear<<<GRID(64 * Ho * Ho), TB, 0, stream>>>(curp, deca, 64, Hc, Hc, Ho, Ho);
    conv3x3<<<GRID(64 * Ho * Ho), TB, 0, stream>>>(
        deca, up_w + (long)i * 64 * 64 * 9, up_b + i * 64, decb, 64, 64, Ho, Ho, 1, 1);
    add_inplace<<<GRID(64 * Ho * Ho), TB, 0, stream>>>(decb, skips[2 - i], 64 * Ho * Ho);
    curp = decb;
    Hc = Ho;
  }
  resize_bilinear<<<GRID(64 * 256 * 256), TB, 0, stream>>>(curp, deca, 64, 64, 64, 256, 256);
  conv3x3<<<GRID(64 * 256 * 256), TB, 0, stream>>>(
      deca, oc_w, oc_b, (float*)d_out, 64, 64, 256, 256, 1, 1);
#undef GRID
}
